// NodeFeatures_14139032338507
// MI455X (gfx1250) — matrix operand order repaired and checked by reading, not compiled, not run on hardware
//
#include <hip/hip_runtime.h>
#include <hip/hip_bf16.h>

typedef _Float16 v16h __attribute__((ext_vector_type(16)));
typedef _Float16 v8h  __attribute__((ext_vector_type(8)));
typedef float    v8f  __attribute__((ext_vector_type(8)));

#define DIM   128
#define NNODE 2048
#define NBAT  8

// ---------------------------------------------------------------------------
// Kernel 1: fp32 -> fp16 conversion of x, Wu, Wv into workspace.
// ---------------------------------------------------------------------------
__global__ void nf_convert_kernel(const float* __restrict__ x,
                                  const float* __restrict__ Wu,
                                  const float* __restrict__ Wv,
                                  _Float16* __restrict__ xh,
                                  _Float16* __restrict__ Wuh,
                                  _Float16* __restrict__ Wvh) {
  int i = blockIdx.x * blockDim.x + threadIdx.x;
  const int nx = NBAT * NNODE * DIM;          // 2097152
  const int nw = DIM * DIM;                   // 16384
  if (i < nx) {
    xh[i] = (_Float16)x[i];
  } else if (i < nx + nw) {
    int j = i - nx;
    Wuh[j] = (_Float16)Wu[j];
  } else {
    int j = i - nx - nw;
    Wvh[j] = (_Float16)Wv[j];
  }
}

// ---------------------------------------------------------------------------
// Kernel 2: Ux = x*Wu^T + bu (fp32, [node,e])  and  VxT = (x*Wv^T + bv)^T
// stored f16 as [b, e, n] so the aggregation GEMM gets contiguous B-fragments.
// One wave computes one 16x16 tile; K = 128 = 4 x WMMA_16x16x32_F16 per output.
// ---------------------------------------------------------------------------
__global__ void nf_linear_kernel(const _Float16* __restrict__ xh,
                                 const _Float16* __restrict__ Wuh,
                                 const _Float16* __restrict__ Wvh,
                                 const float* __restrict__ bu,
                                 const float* __restrict__ bv,
                                 float* __restrict__ Uxf,
                                 _Float16* __restrict__ VxT) {
  const int lane = threadIdx.x & 31;
  const int wid  = (blockIdx.x * blockDim.x + threadIdx.x) >> 5;
  const int mt   = wid >> 3;        // 0..1023 : 16-row tile over B*N
  const int nt   = wid & 7;         // 0..7    : 16-col tile over D
  const int lm   = lane & 15;
  const int half = lane >> 4;
  const int row  = mt * 16 + lm;    // node row for the A fragment
  const int e    = nt * 16 + lm;    // embed column for B / C fragments

  const _Float16* arow  = xh  + (size_t)row * DIM;
  const _Float16* wurow = Wuh + (size_t)e   * DIM;
  const _Float16* wvrow = Wvh + (size_t)e   * DIM;

  v8f cu = {};
  v8f cv = {};
#pragma unroll
  for (int k = 0; k < DIM; k += 32) {
    // A fragment (16x32 f16): lane holds row M=lm, K = half*16 .. half*16+15
    v8h a0 = *(const v8h*)(arow + k + half * 16);
    v8h a1 = *(const v8h*)(arow + k + half * 16 + 8);
    v16h a;
#pragma unroll
    for (int t = 0; t < 8; ++t) { a[t] = a0[t]; a[t + 8] = a1[t]; }
    // B fragment (32x16 f16): lane holds column n=lm, K = half*16 .. half*16+15
    v16h bu_frag = *(const v16h*)(wurow + k + half * 16);
    v16h bv_frag = *(const v16h*)(wvrow + k + half * 16);
    cu = __builtin_amdgcn_wmma_f32_16x16x32_f16(false, a, false, bu_frag,
                                                (short)0, cu, false, false);
    cv = __builtin_amdgcn_wmma_f32_16x16x32_f16(false, a, false, bv_frag,
                                                (short)0, cv, false, false);
  }

  const float biasu = bu[e];
  const float biasv = bv[e];
#pragma unroll
  for (int r = 0; r < 8; ++r) {
    // C layout: VGPR r, lanes 0-15 -> M=r, lanes 16-31 -> M=r+8
    const int node = mt * 16 + r + half * 8;       // global row in [0, B*N)
    Uxf[(size_t)node * DIM + e] = cu[r] + biasu;
    const int b  = node >> 11;                     // node / 2048
    const int nn = node & (NNODE - 1);             // node % 2048
    VxT[((size_t)(b * DIM + e)) * NNODE + nn] = (_Float16)(cv[r] + biasv);
  }
}

// ---------------------------------------------------------------------------
// Kernel 3: agg[b] = mask[b] (2048x2048, {0,1} exact in f16) x Vx[b] (2048x128),
// out = relu(Ux + agg). One BLOCK (8 waves) owns a 16-row strip; each wave takes
// a disjoint 256-wide K slice (mask streamed from HBM exactly once) and computes
// partial sums for all 8 N-tiles; partials are reduced through LDS, then wave w
// finishes N-tile w (Ux add + ReLU + store). 8192 waves for latency hiding.
// ---------------------------------------------------------------------------
__global__ void nf_agg_kernel(const float* __restrict__ mask,
                              const _Float16* __restrict__ VxT,
                              const float* __restrict__ Uxf,
                              float* __restrict__ out) {
  __shared__ float red[8][8][32][8];   // [wave][nt][lane][r] = 64 KB

  const int lane = threadIdx.x & 31;
  const int w    = threadIdx.x >> 5;   // wave id in block: 0..7 (K slice / N-tile)
  const int blk  = blockIdx.x;         // 0..1023
  const int b    = blk >> 7;           // batch
  const int mt   = blk & 127;          // 16-row strip within batch
  const int lm   = lane & 15;
  const int half = lane >> 4;
  const int i    = mt * 16 + lm;       // output row (A fragment row M)

  const float*    maskRow = mask + ((size_t)b * NNODE + i) * NNODE;
  const _Float16* vbase   = VxT + (size_t)b * DIM * NNODE;

  v8f acc[8];
#pragma unroll
  for (int t = 0; t < 8; ++t) acc[t] = (v8f){};

  const int k0 = w * (NNODE / 8);      // this wave's 256-wide K slice
  for (int k = k0; k < k0 + (NNODE / 8); k += 32) {
    const float* mp = maskRow + k + half * 16;
    // stream mask fp32 -> f16 A fragment (exact: values are 0/1)
    float4 m0 = *(const float4*)(mp);
    float4 m1 = *(const float4*)(mp + 4);
    float4 m2 = *(const float4*)(mp + 8);
    float4 m3 = *(const float4*)(mp + 12);
    __builtin_prefetch((const void*)(mp + 32), 0, 0);   // global_prefetch_b8
    v16h a;
    a[0]  = (_Float16)m0.x; a[1]  = (_Float16)m0.y;
    a[2]  = (_Float16)m0.z; a[3]  = (_Float16)m0.w;
    a[4]  = (_Float16)m1.x; a[5]  = (_Float16)m1.y;
    a[6]  = (_Float16)m1.z; a[7]  = (_Float16)m1.w;
    a[8]  = (_Float16)m2.x; a[9]  = (_Float16)m2.y;
    a[10] = (_Float16)m2.z; a[11] = (_Float16)m2.w;
    a[12] = (_Float16)m3.x; a[13] = (_Float16)m3.y;
    a[14] = (_Float16)m3.z; a[15] = (_Float16)m3.w;
#pragma unroll
    for (int nt = 0; nt < 8; ++nt) {
      // B fragment: column e = nt*16+lm of Vx == row of VxT, contiguous 16 halves
      const _Float16* vp = vbase + ((size_t)(nt * 16 + lm)) * NNODE + k + half * 16;
      v16h bf = *(const v16h*)vp;
      acc[nt] = __builtin_amdgcn_wmma_f32_16x16x32_f16(false, a, false, bf,
                                                       (short)0, acc[nt],
                                                       false, false);
    }
  }

  // Stage partial accumulators in LDS (each lane: 8 contiguous floats per tile).
#pragma unroll
  for (int nt = 0; nt < 8; ++nt) {
#pragma unroll
    for (int r = 0; r < 8; ++r) red[w][nt][lane][r] = acc[nt][r];
  }
  __syncthreads();

  // Wave w reduces N-tile nt == w across all 8 K slices, then epilogue.
  const int nt = w;
  v8f tot = {};
#pragma unroll
  for (int ww = 0; ww < 8; ++ww) {
#pragma unroll
    for (int r = 0; r < 8; ++r) tot[r] += red[ww][nt][lane][r];
  }

  const int e = nt * 16 + lm;
#pragma unroll
  for (int r = 0; r < 8; ++r) {
    const int node = b * NNODE + mt * 16 + r + half * 8;
    float v = Uxf[(size_t)node * DIM + e] + tot[r];
    out[(size_t)node * DIM + e] = v > 0.f ? v : 0.f;
  }
}

// ---------------------------------------------------------------------------
// Launcher. Inputs (setup_inputs order): x, mask, Wu, bu, Wv, bv. Output fp32.
// Workspace layout (bytes):
//   [0, 4194304)            xh   : f16 x, [B*N, D]
//   [4194304, +32768)       Wuh  : f16 Wu
//   [4227072, +32768)       Wvh  : f16 Wv
//   [4259840, +4194304)     VxT  : f16 [B, D, N]
//   [8454144, +8388608)     Uxf  : f32 [B*N, D]
// ---------------------------------------------------------------------------
extern "C" void kernel_launch(void* const* d_in, const int* in_sizes, int n_in,
                              void* d_out, int out_size, void* d_ws, size_t ws_size,
                              hipStream_t stream) {
  const float* x    = (const float*)d_in[0];
  const float* mask = (const float*)d_in[1];
  const float* Wu   = (const float*)d_in[2];
  const float* bu   = (const float*)d_in[3];
  const float* Wv   = (const float*)d_in[4];
  const float* bv   = (const float*)d_in[5];
  float* out = (float*)d_out;

  char* ws = (char*)d_ws;
  _Float16* xh  = (_Float16*)(ws);
  _Float16* Wuh = (_Float16*)(ws + 4194304);
  _Float16* Wvh = (_Float16*)(ws + 4227072);
  _Float16* VxT = (_Float16*)(ws + 4259840);
  float*    Uxf = (float*)   (ws + 8454144);

  // 1) convert: (B*N*D + 2*D*D) = 2129920 elements = 8320 * 256
  nf_convert_kernel<<<8320, 256, 0, stream>>>(x, Wu, Wv, xh, Wuh, Wvh);

  // 2) linear: 1024 m-tiles * 8 n-tiles = 8192 waves = 1024 blocks * 8 waves
  nf_linear_kernel<<<1024, 256, 0, stream>>>(xh, Wuh, Wvh, bu, bv, Uxf, VxT);

  // 3) aggregation + epilogue: 1024 strips, one block (8 waves) each = 8192 waves
  nf_agg_kernel<<<1024, 256, 0, stream>>>(mask, VxT, Uxf, out);
}
